// QRNN_19662360281095
// MI455X (gfx1250) — matrix operand order repaired and checked by reading, not compiled, not run on hardware
//
#include <hip/hip_runtime.h>
#include <cmath>

// ---------------- types ----------------
typedef __bf16 v16bf __attribute__((ext_vector_type(16)));
typedef __bf16 v8bf  __attribute__((ext_vector_type(8)));
typedef float  v8f   __attribute__((ext_vector_type(8)));
typedef float  v4f   __attribute__((ext_vector_type(4)));

// ---------------- problem constants ----------------
#define B_  8
#define T_  4096
#define D_  512
#define H_  512
#define M_  (B_ * T_)      // 32768 rows
#define N3_ (3 * H_)       // 1536 cols
#define K_  512            // contraction (D_ or H_)

// GEMM tiling
#define BM 128
#define BN 128
#define BK 32
#define LDA_ 40            // padded LDS stride (elements) -> conflict-free b128 reads
#define LDB_ 40

// scan chunking
#define NC  32             // chunks along T
#define LCH (T_ / NC)      // 128 steps per chunk

// ---------------- CDNA5 async-copy helpers ----------------
__device__ __forceinline__ unsigned lds_off(const void* p) {
  // generic LDS pointer = {SHARED_BASE aperture, 32-bit LDS offset}
  return (unsigned)(uintptr_t)p;
}
__device__ __forceinline__ void async_b128(unsigned lds_byte, const void* gptr) {
  asm volatile("global_load_async_to_lds_b128 %0, %1, off"
               :: "v"(lds_byte), "v"(gptr) : "memory");
}
__device__ __forceinline__ void wait_async_le4() {
  asm volatile("s_wait_asynccnt 0x4" ::: "memory");
}
__device__ __forceinline__ void wait_async_le0() {
  asm volatile("s_wait_asynccnt 0x0" ::: "memory");
}

// =====================================================================
// f32 -> bf16 conversion pre-pass (8 elements / thread)
// =====================================================================
__global__ __launch_bounds__(256)
void cvt_f32_to_bf16(const float* __restrict__ in, __bf16* __restrict__ out, int n) {
  int i = (blockIdx.x * 256 + threadIdx.x) * 8;
  if (i >= n) return;
  v4f a = *(const v4f*)(in + i);
  v4f b = *(const v4f*)(in + i + 4);
  v8bf o;
  for (int e = 0; e < 4; ++e) { o[e] = (__bf16)a[e]; o[4 + e] = (__bf16)b[e]; }
  *(v8bf*)(out + i) = o;
}

// =====================================================================
// GEMM: Y[M,N] = X[M,K] * W[N,K]^T + bias[N]; bf16 inputs, f32 out.
// 256 threads / 8 waves; wave (wm,wn) owns 32x64; 2x4 16x16 f32 accums.
// Global->LDS staging uses async-to-LDS b128 with double buffering.
// =====================================================================
__global__ __launch_bounds__(256)
void qrnn_gemm_bias(const __bf16* __restrict__ X, const __bf16* __restrict__ W,
                    const float* __restrict__ bias, float* __restrict__ Y,
                    int M, int N, int K) {
  __shared__ __align__(16) __bf16 As[2][BM * LDA_];
  __shared__ __align__(16) __bf16 Bs[2][BN * LDB_];

  const int tid  = threadIdx.x;
  const int lane = tid & 31;
  const int wave = tid >> 5;
  const int wm   = wave & 3;    // 0..3 -> 32-row slice
  const int wn   = wave >> 2;   // 0..1 -> 64-col slice
  const int bm   = blockIdx.y * BM;
  const int bn   = blockIdx.x * BN;
  const int hi   = lane >> 4;   // K-half selector (wave32 WMMA layout)
  const int lo   = lane & 15;

  union Frag { v16bf v; v8bf h[2]; };

  const unsigned aoff[2] = { lds_off(&As[0][0]), lds_off(&As[1][0]) };
  const unsigned boff[2] = { lds_off(&Bs[0][0]), lds_off(&Bs[1][0]) };

  // stage one K-tile (128x32 bf16 A + B) via 4 async b128 ops per lane
  auto stage = [&](int k0, int buf) {
    for (int i = 0; i < 2; ++i) {
      int idx = tid + i * 256;        // 0..511 chunk id
      int r   = idx >> 2;             // 0..127 row
      int c8  = (idx & 3) * 8;        // 0,8,16,24 (elements)
      async_b128(aoff[buf] + (unsigned)(r * LDA_ + c8) * 2,
                 X + (size_t)(bm + r) * K + k0 + c8);
      async_b128(boff[buf] + (unsigned)(r * LDB_ + c8) * 2,
                 W + (size_t)(bn + r) * K + k0 + c8);
    }
  };

  // accumulators initialized with the bias (broadcast over the 8 row-VGPRs)
  v8f acc[2][4];
  for (int nt = 0; nt < 4; ++nt) {
    float bv = bias[bn + wn * 64 + nt * 16 + lo];
    for (int mt = 0; mt < 2; ++mt)
      for (int r = 0; r < 8; ++r) acc[mt][nt][r] = bv;
  }

  stage(0, 0);
  int buf = 0;
  for (int k0 = 0; k0 < K; k0 += BK) {
    const bool nxt = (k0 + BK) < K;
    if (nxt) stage(k0 + BK, buf ^ 1);     // keep next tile in flight
    if (nxt) wait_async_le4(); else wait_async_le0();
    __syncthreads();                       // this buffer's tile is resident

    // ---- fragments per ISA wave32 layouts ----
    // A (16x32): lanes 0-15 row m: K0..7 @+0, K16..23 @+16;
    //            lanes16-31:       K8..15 @+8, K24..31 @+24.
    Frag a[2];
    for (int mt = 0; mt < 2; ++mt) {
      int m = wm * 32 + mt * 16 + lo;
      a[mt].h[0] = *(const v8bf*)(&As[buf][m * LDA_ + hi * 8]);
      a[mt].h[1] = *(const v8bf*)(&As[buf][m * LDA_ + 16 + hi * 8]);
    }
    // B (32x16): lanes 0-15 col n: K0..7 @+0, K16..23 @+16;
    //            lanes16-31:       K8..15 @+8, K24..31 @+24.
    Frag b[4];
    for (int nt = 0; nt < 4; ++nt) {
      int n = wn * 64 + nt * 16 + lo;
      b[nt].h[0] = *(const v8bf*)(&Bs[buf][n * LDB_ + hi * 8]);
      b[nt].h[1] = *(const v8bf*)(&Bs[buf][n * LDB_ + 16 + hi * 8]);
    }

    // ---- 8 WMMAs per wave per K-step ----
    for (int mt = 0; mt < 2; ++mt)
      for (int nt = 0; nt < 4; ++nt)
        acc[mt][nt] = __builtin_amdgcn_wmma_f32_16x16x32_bf16(
            false, a[mt].v, false, b[nt].v, (short)0, acc[mt][nt], false, false);

    __syncthreads();                       // all waves done reading this buffer
    buf ^= 1;
  }

  // ---- epilogue: C/D layout -> row = base + r + hi*8, col = base + lo ----
  for (int mt = 0; mt < 2; ++mt) {
    int rowb = bm + wm * 32 + mt * 16 + hi * 8;
    for (int nt = 0; nt < 4; ++nt) {
      int col = bn + wn * 64 + nt * 16 + lo;
      for (int r = 0; r < 8; ++r)
        Y[(size_t)(rowb + r) * N + col] = acc[mt][nt][r];
    }
  }
}

// =====================================================================
// Chunked linear-recurrence scan: c_t = s*tanh(z) + (1-s)*c_{t-1},
// s = sigmoid(f). Chunk map is affine: c_out = Ce + Ac * c_in.
// =====================================================================
__device__ __forceinline__ float sigf(float x) { return 1.0f / (1.0f + __expf(-x)); }

// phase 1: per-chunk affine map (A_j, c_end_j). tid = (b*NC + j)*H + h.
__global__ __launch_bounds__(256)
void qrnn_scan_phase1(const float* __restrict__ y, float* __restrict__ Ac,
                      float* __restrict__ Ce) {
  int tid = blockIdx.x * 256 + threadIdx.x;       // B_*NC*H_ threads
  int h  = tid % H_;
  int bj = tid / H_;
  int b  = bj / NC;
  int j  = bj % NC;
  size_t base = ((size_t)b * T_ + (size_t)j * LCH) * N3_ + h;
  float c = 0.0f, A = 1.0f;
  for (int t = 0; t < LCH; ++t) {
    float z = y[base];
    float f = y[base + H_];
    float s = sigf(f);
    float aa = 1.0f - s;
    c = s * tanhf(z) + aa * c;
    A *= aa;
    base += N3_;
  }
  Ac[tid] = A;
  Ce[tid] = c;
}

// phase 2: combine 32 chunk maps per channel; also emits final hidden state.
__global__ __launch_bounds__(256)
void qrnn_scan_phase2(const float* __restrict__ Ac, const float* __restrict__ Ce,
                      float* __restrict__ carry, float* __restrict__ hid) {
  int tid = blockIdx.x * 256 + threadIdx.x;       // B_*H_ threads
  int h = tid % H_;
  int b = tid / H_;
  float c = 0.0f;
  for (int j = 0; j < NC; ++j) {
    int idx = (b * NC + j) * H_ + h;
    carry[idx] = c;
    c = Ce[idx] + Ac[idx] * c;
  }
  hid[b * H_ + h] = c;                            // c_{T-1}
}

// phase 3: replay each chunk from its carry, write gated output sigmoid(o)*c.
// OutT = __bf16 for layer-0 (feeds the next WMMA GEMM), float for layer-1.
template <typename OutT>
__global__ __launch_bounds__(256)
void qrnn_scan_phase3(const float* __restrict__ y, const float* __restrict__ carry,
                      OutT* __restrict__ out) {
  int tid = blockIdx.x * 256 + threadIdx.x;       // B_*NC*H_ threads
  int h  = tid % H_;
  int bj = tid / H_;
  int b  = bj / NC;
  int j  = bj % NC;
  size_t base  = ((size_t)b * T_ + (size_t)j * LCH) * N3_ + h;
  size_t obase = ((size_t)b * T_ + (size_t)j * LCH) * H_ + h;
  float c = carry[tid];
  for (int t = 0; t < LCH; ++t) {
    float z = y[base];
    float f = y[base + H_];
    float o = y[base + 2 * H_];
    float s = sigf(f);
    c = s * tanhf(z) + (1.0f - s) * c;
    out[obase] = (OutT)(sigf(o) * c);
    base += N3_;
    obase += H_;
  }
}

// =====================================================================
// launcher
// =====================================================================
extern "C" void kernel_launch(void* const* d_in, const int* in_sizes, int n_in,
                              void* d_out, int out_size, void* d_ws, size_t ws_size,
                              hipStream_t stream) {
  const float* x  = (const float*)d_in[0];   // [B,T,D]
  const float* W0 = (const float*)d_in[1];   // [3H,D]
  const float* b0 = (const float*)d_in[2];   // [3H]
  const float* W1 = (const float*)d_in[3];   // [3H,H]
  const float* b1 = (const float*)d_in[4];   // [3H]
  float* out = (float*)d_out;                // [B,T,H] ++ [2,B,H]

  char* ws = (char*)d_ws;
  size_t off = 0;
  float* y = (float*)(ws + off);  off += (size_t)M_ * N3_ * sizeof(float);   // 192 MB
  __bf16* xb  = (__bf16*)(ws + off); off += (size_t)M_ * D_ * sizeof(__bf16); // 32 MB
  __bf16* o0b = (__bf16*)(ws + off); off += (size_t)M_ * H_ * sizeof(__bf16); // 32 MB
  __bf16* w0b = (__bf16*)(ws + off); off += (size_t)N3_ * D_ * sizeof(__bf16);
  __bf16* w1b = (__bf16*)(ws + off); off += (size_t)N3_ * H_ * sizeof(__bf16);
  float* Ac = (float*)(ws + off); off += (size_t)B_ * H_ * NC * sizeof(float);
  float* Ce = (float*)(ws + off); off += (size_t)B_ * H_ * NC * sizeof(float);
  float* cy = (float*)(ws + off); off += (size_t)B_ * H_ * NC * sizeof(float);

  float* hid0 = out + (size_t)B_ * T_ * H_;       // hids[0]
  float* hid1 = hid0 + (size_t)B_ * H_;           // hids[1]

  dim3 blk(256);
  dim3 ggemm(N3_ / BN, M_ / BM);                  // (12, 256)
  int  gscan = (B_ * NC * H_) / 256;              // 512 blocks
  int  gcomb = (B_ * H_) / 256;                   // 16 blocks

  // ---- precision pre-pass: f32 -> bf16 for WMMA operands ----
  cvt_f32_to_bf16<<<(M_ * D_) / 2048, blk, 0, stream>>>(x, xb, M_ * D_);
  cvt_f32_to_bf16<<<(N3_ * D_) / 2048, blk, 0, stream>>>(W0, w0b, N3_ * D_);
  cvt_f32_to_bf16<<<(N3_ * H_) / 2048, blk, 0, stream>>>(W1, w1b, N3_ * H_);

  // ---- layer 0 ----
  qrnn_gemm_bias<<<ggemm, blk, 0, stream>>>(xb, w0b, b0, y, M_, N3_, K_);
  qrnn_scan_phase1<<<gscan, blk, 0, stream>>>(y, Ac, Ce);
  qrnn_scan_phase2<<<gcomb, blk, 0, stream>>>(Ac, Ce, cy, hid0);
  qrnn_scan_phase3<__bf16><<<gscan, blk, 0, stream>>>(y, cy, o0b);

  // ---- layer 1 ----
  qrnn_gemm_bias<<<ggemm, blk, 0, stream>>>(o0b, w1b, b1, y, M_, N3_, K_);
  qrnn_scan_phase1<<<gscan, blk, 0, stream>>>(y, Ac, Ce);
  qrnn_scan_phase2<<<gcomb, blk, 0, stream>>>(Ac, Ce, cy, hid1);
  qrnn_scan_phase3<float><<<gscan, blk, 0, stream>>>(y, cy, out);
}
